// RNN_33775622816124
// MI455X (gfx1250) — compile-verified, matrix operand order checked by reading, not run on hardware
//
#include <hip/hip_runtime.h>

// ---------------------------------------------------------------------------
// Stacked ReLU RNN  (B=128, T=784, I=1, H=256, L=20, C=10) for gfx1250.
//  - activations kept as bf16 (halves HBM traffic, feeds WMMA directly)
//  - projection GEMM: v_wmma_f32_16x16x32_bf16, 16x64 tile per wave
//  - recurrent scan: 8 persistent WGs, W_hh slice held in VGPRs,
//    h tile ping-pongs through LDS, 1 barrier per timestep,
//    4 independent WMMA chains of length 4 to hide bf16 WMMA RAW latency
// ---------------------------------------------------------------------------

#define RNN_B 128
#define RNN_T 784
#define RNN_H 256
#define RNN_L 20
#define RNN_C 10
#define RNN_M (RNN_B * RNN_T)          // 100352 rows

typedef __bf16 v16bf __attribute__((ext_vector_type(16)));
typedef float  v8f   __attribute__((ext_vector_type(8)));

union FragBF { unsigned int u[8]; v16bf v; };

__device__ __forceinline__ unsigned short f2bf(float f) {
    union { float f; unsigned int u; } cv; cv.f = f;
    unsigned int u = cv.u;
    u += 0x7FFFu + ((u >> 16) & 1u);   // round-to-nearest-even
    return (unsigned short)(u >> 16);
}

// Load one 16x32 bf16 WMMA fragment (A or B operand) from a row-major tile.
// Documented CDNA5 16-bit layout: lane = row&15, hi half of wave takes the
// +8 K offset inside each 16-wide K group; VGPR v holds K pair
//   k = (v>>2)*16 + hi*8 + (v&3)*2.
// Per lane this is two contiguous 16B runs -> coalesces to 2x b128 loads.
__device__ __forceinline__ v16bf load_frag(const unsigned short* p, int ldk,
                                           int k0, int lane) {
    const int r  = lane & 15;
    const int hi = (lane >> 4) & 1;
    const unsigned short* row = p + r * ldk + k0;
    FragBF f;
#pragma unroll
    for (int v = 0; v < 8; ++v) {
        const int kk = ((v >> 2) << 4) + (hi << 3) + ((v & 3) << 1);
        f.u[v] = *(const unsigned int*)(row + kk);
    }
    return f.v;
}

__device__ __forceinline__ v8f wmma_bf16(v16bf a, v16bf b, v8f c) {
    return __builtin_amdgcn_wmma_f32_16x16x32_bf16(
        /*neg_a=*/false, a, /*neg_b=*/false, b,
        /*c_mod=*/(short)0, c, /*reuse_a=*/false, /*reuse_b=*/false);
}

// --------------------------- weight conversion -----------------------------
__global__ void convert_w_kernel(const float* __restrict__ wih,
                                 const float* __restrict__ whh,
                                 unsigned short* __restrict__ wih_b,
                                 unsigned short* __restrict__ whh_b,
                                 int n_ih, int n_hh) {
    int i = blockIdx.x * blockDim.x + threadIdx.x;
    const int total = n_ih + n_hh;
    for (; i < total; i += gridDim.x * blockDim.x) {
        if (i < n_ih) wih_b[i] = f2bf(wih[i]);
        else          whh_b[i - n_ih] = f2bf(whh[i - n_ih]);
    }
}

// --------------------------- layer 0 projection ----------------------------
// I == 1, so xp[b,t,h] = x[b,t] * W_ih0[h] + b_ih0[h].  float4 vectorized.
__global__ void layer0_xp_kernel(const float* __restrict__ x,
                                 const float* __restrict__ w,
                                 const float* __restrict__ b,
                                 float* __restrict__ xp, int n4) {
    int i = blockIdx.x * blockDim.x + threadIdx.x;
    if (i >= n4) return;
    const int c4 = (i & 63) << 2;       // column group (H/4 = 64 per row)
    const int bt = i >> 6;
    const float  xv = x[bt];
    const float4 wv = *(const float4*)(w + c4);
    const float4 bv = *(const float4*)(b + c4);
    float4 o;
    o.x = xv * wv.x + bv.x;
    o.y = xv * wv.y + bv.y;
    o.z = xv * wv.z + bv.z;
    o.w = xv * wv.w + bv.w;
    *(float4*)(xp + (size_t)bt * RNN_H + c4) = o;
}

// --------------------------- projection GEMM -------------------------------
// xp[M,256] = act[M,256](bf16) @ W[256,256]^T (bf16) + bias, fp32 out.
// Block = 8 waves; wave -> 16 rows x 64 cols; grid = M/32 blocks.
__global__ void __launch_bounds__(256) gemm_xp_kernel(
    const unsigned short* __restrict__ A,   // [M, 256] bf16
    const unsigned short* __restrict__ Wb,  // [256, 256] bf16 (row n = out col n)
    const float* __restrict__ bias,         // [256]
    float* __restrict__ xp)                 // [M, 256] f32
{
    const int lane = threadIdx.x & 31;
    const int wave = threadIdx.x >> 5;
    const int mt   = blockIdx.x * 2 + (wave >> 2);  // 16-row tile
    const int nt   = wave & 3;                      // 64-col group

    const unsigned short* Abase = A + (size_t)mt * 16 * RNN_H;
    v8f acc[4] = {};

#pragma unroll
    for (int k0 = 0; k0 < RNN_H; k0 += 32) {
        __builtin_prefetch(Abase + (lane & 15) * RNN_H + k0 + 64, 0, 0);
        const v16bf a = load_frag(Abase, RNN_H, k0, lane);
#pragma unroll
        for (int j = 0; j < 4; ++j) {
            const v16bf b =
                load_frag(Wb + (size_t)(nt * 64 + j * 16) * RNN_H, RNN_H, k0, lane);
            acc[j] = wmma_bf16(a, b, acc[j]);
        }
    }

    const int hi = (lane >> 4) & 1, nn = lane & 15;
#pragma unroll
    for (int j = 0; j < 4; ++j) {
        const int c = nt * 64 + j * 16 + nn;
        const float bj = bias[c];
#pragma unroll
        for (int r = 0; r < 8; ++r) {
            const int m = mt * 16 + r + (hi << 3);
            xp[(size_t)m * RNN_H + c] = acc[j][r] + bj;
        }
    }
}

// --------------------------- recurrent scan --------------------------------
// 8 blocks (one per 16-row B tile), 8 waves each (wave -> 32 cols).
// W_hh slice lives in VGPRs; h tile ping-pongs through LDS.
// Per step: xp loads issued first (overlap WMMA chain), then 16 WMMAs
// arranged as 4 independent chains of 4 (hides 5-slot bf16 RAW hazard).
__global__ void __launch_bounds__(256) scan_kernel(
    const float* __restrict__ xp,            // [B*T, H] f32 (incl. b_ih)
    const unsigned short* __restrict__ Whh,  // [H, H] bf16 row-major
    const float* __restrict__ bhh,           // [H]
    unsigned short* __restrict__ act,        // [B*T, H] bf16 out
    float* __restrict__ final_h)             // [B, H] f32 (t = T-1)
{
    __shared__ unsigned short hbuf[2][16 * RNN_H];

    const int lane = threadIdx.x & 31;
    const int wave = threadIdx.x >> 5;
    const int bt0  = blockIdx.x * 16;        // first batch row of this tile
    const int col0 = wave * 32;              // this wave's 32 output cols
    const int hi   = (lane >> 4) & 1, nn = lane & 15;

    // Preload this wave's W_hh slice (32 rows x 256 K) into registers:
    // 2 col-tiles x 8 K-steps x 8 VGPRs = 128 VGPRs.
    v16bf bfrag[2][8];
#pragma unroll
    for (int j = 0; j < 2; ++j)
#pragma unroll
        for (int kk = 0; kk < 8; ++kk)
            bfrag[j][kk] =
                load_frag(Whh + (size_t)(col0 + j * 16) * RNN_H, RNN_H, kk * 32, lane);

    const float bias0 = bhh[col0 + nn];
    const float bias1 = bhh[col0 + 16 + nn];

    for (int i = threadIdx.x; i < 16 * RNN_H; i += 256) hbuf[0][i] = 0;
    __syncthreads();

#pragma unroll 1
    for (int t = 0; t < RNN_T; ++t) {
        const unsigned short* hread  = hbuf[t & 1];
        unsigned short*       hwrite = hbuf[(t + 1) & 1];

        if (t + 8 < RNN_T)
            __builtin_prefetch(xp + ((size_t)(bt0 + nn) * RNN_T + (t + 8)) * RNN_H, 0, 0);

        // Issue this step's xp loads up front so they overlap the WMMA chain.
        float  xv0[8], xv1[8];
        size_t rowg[8];
#pragma unroll
        for (int r = 0; r < 8; ++r) {
            const int m = r + (hi << 3);
            rowg[r] = ((size_t)(bt0 + m) * RNN_T + t) * RNN_H;
            xv0[r] = xp[rowg[r] + col0 + nn];
            xv1[r] = xp[rowg[r] + col0 + 16 + nn];
        }

        // 4 independent accumulator chains (2 col-tiles x 2 K-halves).
        v8f a0 = {}, a1 = {}, b0 = {}, b1 = {};
#pragma unroll
        for (int kk = 0; kk < 4; ++kk) {
            const v16bf x0 = load_frag(hread, RNN_H, kk * 32, lane);
            const v16bf x1 = load_frag(hread, RNN_H, (kk + 4) * 32, lane);
            a0 = wmma_bf16(x0, bfrag[0][kk],     a0);
            a1 = wmma_bf16(x0, bfrag[1][kk],     a1);
            b0 = wmma_bf16(x1, bfrag[0][kk + 4], b0);
            b1 = wmma_bf16(x1, bfrag[1][kk + 4], b1);
        }

#pragma unroll
        for (int r = 0; r < 8; ++r) {
            const int m = r + (hi << 3);
            {
                const int c = col0 + nn;
                float v = (a0[r] + b0[r]) + xv0[r] + bias0;
                v = v > 0.f ? v : 0.f;
                const unsigned short bv = f2bf(v);
                hwrite[m * RNN_H + c] = bv;
                act[rowg[r] + c] = bv;
                if (t == RNN_T - 1) final_h[(size_t)(bt0 + m) * RNN_H + c] = v;
            }
            {
                const int c = col0 + 16 + nn;
                float v = (a1[r] + b1[r]) + xv1[r] + bias1;
                v = v > 0.f ? v : 0.f;
                const unsigned short bv = f2bf(v);
                hwrite[m * RNN_H + c] = bv;
                act[rowg[r] + c] = bv;
                if (t == RNN_T - 1) final_h[(size_t)(bt0 + m) * RNN_H + c] = v;
            }
        }
        __syncthreads();
    }
}

// ------------------------------- final FC ----------------------------------
__global__ void fc_kernel(const float* __restrict__ h,
                          const float* __restrict__ Wfc,
                          const float* __restrict__ bfc,
                          float* __restrict__ out) {
    const int i = blockIdx.x * blockDim.x + threadIdx.x;
    if (i >= RNN_B * RNN_C) return;
    const int b = i / RNN_C, c = i % RNN_C;
    const float* hr = h + (size_t)b * RNN_H;
    const float* wr = Wfc + (size_t)c * RNN_H;
    float s = bfc[c];
#pragma unroll 8
    for (int k = 0; k < RNN_H; ++k) s += hr[k] * wr[k];
    out[i] = s;
}

// ------------------------------- launcher ----------------------------------
extern "C" void kernel_launch(void* const* d_in, const int* in_sizes, int n_in,
                              void* d_out, int out_size, void* d_ws, size_t ws_size,
                              hipStream_t stream) {
    (void)in_sizes; (void)n_in; (void)out_size; (void)ws_size;

    const float* x     = (const float*)d_in[0];
    const float* W_ih0 = (const float*)d_in[1];
    const float* b_ih0 = (const float*)d_in[2];
    const float* W_ih  = (const float*)d_in[3];   // [L-1, H, H]
    const float* b_ih  = (const float*)d_in[4];   // [L-1, H]
    const float* W_hh  = (const float*)d_in[5];   // [L, H, H]
    const float* b_hh  = (const float*)d_in[6];   // [L, H]
    const float* W_fc  = (const float*)d_in[7];
    const float* b_fc  = (const float*)d_in[8];

    char* ws = (char*)d_ws;
    const size_t xp_bytes  = (size_t)RNN_M * RNN_H * sizeof(float);          // 102.8 MB
    const size_t act_bytes = (size_t)RNN_M * RNN_H * sizeof(unsigned short); //  51.4 MB
    const size_t wih_elems = (size_t)(RNN_L - 1) * RNN_H * RNN_H;
    const size_t whh_elems = (size_t)RNN_L * RNN_H * RNN_H;

    float*          xp    = (float*)ws;
    unsigned short* act   = (unsigned short*)(ws + xp_bytes);
    unsigned short* wih_b = (unsigned short*)(ws + xp_bytes + act_bytes);
    unsigned short* whh_b = wih_b + wih_elems;
    float*          fin   = (float*)((char*)(whh_b + whh_elems));
    float*          out   = (float*)d_out;

    // 1. weights -> bf16 (once; tiny)
    {
        const int total = (int)(wih_elems + whh_elems);
        convert_w_kernel<<<(total + 255) / 256, 256, 0, stream>>>(
            W_ih, W_hh, wih_b, whh_b, (int)wih_elems, (int)whh_elems);
    }

    // 2. layer 0 input projection (I == 1 -> elementwise)
    {
        const int n4 = RNN_M * (RNN_H / 4);
        layer0_xp_kernel<<<(n4 + 255) / 256, 256, 0, stream>>>(x, W_ih0, b_ih0, xp, n4);
    }

    // 3. layer 0 scan
    scan_kernel<<<RNN_B / 16, 256, 0, stream>>>(xp, whh_b, b_hh, act, fin);

    // 4. layers 1..L-1: projection GEMM then scan
    for (int l = 1; l < RNN_L; ++l) {
        gemm_xp_kernel<<<RNN_M / 32, 256, 0, stream>>>(
            act, wih_b + (size_t)(l - 1) * RNN_H * RNN_H, b_ih + (size_t)(l - 1) * RNN_H, xp);
        scan_kernel<<<RNN_B / 16, 256, 0, stream>>>(
            xp, whh_b + (size_t)l * RNN_H * RNN_H, b_hh + (size_t)l * RNN_H, act, fin);
    }

    // 5. final FC
    fc_kernel<<<(RNN_B * RNN_C + 255) / 256, 256, 0, stream>>>(fin, W_fc, b_fc, out);
}
